// FlowsheetGNN_61830349193982
// MI455X (gfx1250) — compile-verified, matrix operand order checked by reading, not run on hardware
//
#include <hip/hip_runtime.h>
#include <hip/hip_bf16.h>

#define NN 100000
#define NE 1600000
#define HID 128
#define NG 64

typedef __attribute__((ext_vector_type(16))) __bf16 v16bf;
typedef __attribute__((ext_vector_type(8)))  float  v8f;

__device__ __forceinline__ void atomAddF(float* p, float v) {
    __hip_atomic_fetch_add(p, v, __ATOMIC_RELAXED, __HIP_MEMORY_SCOPE_AGENT);
}

__device__ __forceinline__ unsigned packbf(float a, float b) {
    union { __bf16 h[2]; unsigned u; } c;
    c.h[0] = (__bf16)a; c.h[1] = (__bf16)b;
    return c.u;
}

// ---------------- node embedding: h = x @ Wne + bne ----------------
__global__ __launch_bounds__(128)
void node_embed_kernel(const float* __restrict__ x, const float* __restrict__ w,
                       const float* __restrict__ b, float* __restrict__ h) {
    int n = blockIdx.x, j = threadIdx.x;
    const float* xr = x + n * 12;
    float acc = b[j];
#pragma unroll
    for (int k = 0; k < 12; ++k) acc += xr[k] * w[k * HID + j];
    h[n * HID + j] = acc;
}

// ---------------- z = (1+eps) * h  (vectorized) ----------------
__global__ __launch_bounds__(256)
void copy_scale_kernel(const float4* __restrict__ h, float4* __restrict__ z,
                       const float* __restrict__ epsp, int n4) {
    int i = blockIdx.x * 256 + threadIdx.x;
    if (i >= n4) return;
    float s = 1.0f + epsp[0];
    float4 v = h[i];
    v.x *= s; v.y *= s; v.z *= s; v.w *= s;
    z[i] = v;
}

__global__ __launch_bounds__(256)
void zero_kernel(float* __restrict__ p, int n) {
    int i = blockIdx.x * 256 + threadIdx.x;
    if (i < n) p[i] = 0.0f;
}

// ------- edge message: z[dst] += relu(h[src] + edge_attr@We + be) -------
__global__ __launch_bounds__(256)
void message_kernel(const float* __restrict__ h, const int* __restrict__ ei,
                    const float* __restrict__ ea, const float* __restrict__ we,
                    const float* __restrict__ be, float* __restrict__ z) {
    int sub = threadIdx.x >> 7;          // 2 edges per block
    int j   = threadIdx.x & 127;
    long e  = (long)blockIdx.x * 2 + sub;
    if (e >= NE) return;
    int src = ei[e];
    int dst = ei[NE + e];
    const float* ar = ea + e * 10;
    float acc = be[j];
#pragma unroll
    for (int k = 0; k < 10; ++k) acc += ar[k] * we[k * HID + j];
    float m = h[(long)src * HID + j] + acc;
    m = m > 0.0f ? m : 0.0f;
    atomAddF(&z[(long)dst * HID + j], m);
}

// ------- fused MLP: z <- relu(z@W1+b1)@W2+b2 (in place) + BN stats -------
// One wave owns 16 rows. 8 waves/block -> 128 rows/block.
__global__ __launch_bounds__(256)
void gemm_mlp_kernel(float* __restrict__ z,
                     const float* __restrict__ w1, const float* __restrict__ b1,
                     const float* __restrict__ w2, const float* __restrict__ b2,
                     float* __restrict__ ssum, float* __restrict__ ssum2) {
    __shared__ unsigned w1p[HID * 64];        // [col][kpair] bf16x2, 32 KB
    __shared__ unsigned w2p[HID * 64];        // 32 KB
    __shared__ __bf16   tmid[8][16][HID];     // per-wave intermediate, 32 KB

    const int tid = threadIdx.x;
    // pack both weight matrices into LDS as B-fragment-ready bf16 pairs
    for (int e = tid; e < HID * 64; e += 256) {
        int col = e >> 6, kk = (e & 63) << 1;
        w1p[e] = packbf(w1[kk * HID + col], w1[(kk + 1) * HID + col]);
        w2p[e] = packbf(w2[kk * HID + col], w2[(kk + 1) * HID + col]);
    }
    __syncthreads();

    const int wave  = tid >> 5, lane = tid & 31;
    const int mrow  = lane & 15;
    const int khalf = (lane < 16) ? 0 : 8;     // A/B fragment K-half per ISA layout
    const int row0  = blockIdx.x * 128 + wave * 16;
    const int grow  = row0 + mrow;
    const bool valid = grow < NN;
    const float* zr = z + (long)grow * HID;

    // -------- load full 16x128 A-tile (fp32 -> bf16 fragments) --------
    v16bf afr[4];
#pragma unroll
    for (int ks = 0; ks < 4; ++ks) {
#pragma unroll
        for (int v = 0; v < 8; ++v) {
            int kk = ks * 32 + khalf + ((v < 4) ? (2 * v) : (16 + 2 * (v - 4)));
            float f0 = 0.0f, f1 = 0.0f;
            if (valid) { f0 = zr[kk]; f1 = zr[kk + 1]; }
            afr[ks][2 * v]     = (__bf16)f0;
            afr[ks][2 * v + 1] = (__bf16)f1;
        }
    }

    // -------- GEMM1: t = relu(z@W1 + b1) -> wave-private LDS --------
    const int rbase = (lane < 16) ? 0 : 8;
#pragma unroll
    for (int ct = 0; ct < 8; ++ct) {
        v8f acc = {};
#pragma unroll
        for (int ks = 0; ks < 4; ++ks) {
            v16bf bfr;
            const unsigned* wp = &w1p[(ct * 16 + mrow) * 64 + ks * 16 + (khalf >> 1)];
#pragma unroll
            for (int v = 0; v < 8; ++v) {
                union { unsigned u; __bf16 h[2]; } c;
                c.u = wp[(v < 4) ? v : (v + 4)];
                bfr[2 * v] = c.h[0]; bfr[2 * v + 1] = c.h[1];
            }
            acc = __builtin_amdgcn_wmma_f32_16x16x32_bf16(
                false, afr[ks], false, bfr, (short)0, acc, false, false);
        }
        int coln = ct * 16 + mrow;
        float bb = b1[coln];
#pragma unroll
        for (int r = 0; r < 8; ++r) {
            float vv = acc[r] + bb;
            vv = vv > 0.0f ? vv : 0.0f;
            tmid[wave][rbase + r][coln] = (__bf16)vv;
        }
    }
    // wave-private LDS region; same-wave DS ops are in-order -> no barrier needed

    // -------- GEMM2: z = t@W2 + b2 (in place) + BN partial stats --------
#pragma unroll
    for (int ct = 0; ct < 8; ++ct) {
        v8f acc = {};
#pragma unroll
        for (int ks = 0; ks < 4; ++ks) {
            v16bf a2;
            const __bf16* tr = &tmid[wave][mrow][0];
#pragma unroll
            for (int v = 0; v < 8; ++v) {
                int kk = ks * 32 + khalf + ((v < 4) ? (2 * v) : (16 + 2 * (v - 4)));
                a2[2 * v] = tr[kk]; a2[2 * v + 1] = tr[kk + 1];
            }
            v16bf bfr;
            const unsigned* wp = &w2p[(ct * 16 + mrow) * 64 + ks * 16 + (khalf >> 1)];
#pragma unroll
            for (int v = 0; v < 8; ++v) {
                union { unsigned u; __bf16 h[2]; } c;
                c.u = wp[(v < 4) ? v : (v + 4)];
                bfr[2 * v] = c.h[0]; bfr[2 * v + 1] = c.h[1];
            }
            acc = __builtin_amdgcn_wmma_f32_16x16x32_bf16(
                false, a2, false, bfr, (short)0, acc, false, false);
        }
        int coln = ct * 16 + mrow;
        float bb = b2[coln];
        float s = 0.0f, s2 = 0.0f;
#pragma unroll
        for (int r = 0; r < 8; ++r) {
            float vv = acc[r] + bb;
            int gr = row0 + rbase + r;
            if (gr < NN) {
                z[(long)gr * HID + coln] = vv;
                s += vv; s2 += vv * vv;
            }
        }
        atomAddF(&ssum[coln], s);
        atomAddF(&ssum2[coln], s2);
    }
}

// ---------------- BN finalize: scale/shift per feature ----------------
__global__ __launch_bounds__(128)
void bn_final_kernel(const float* __restrict__ ssum, const float* __restrict__ ssum2,
                     const float* __restrict__ gamma, const float* __restrict__ beta,
                     float* __restrict__ scale, float* __restrict__ shift) {
    int j = threadIdx.x;
    float mu  = ssum[j]  * (1.0f / NN);
    float var = ssum2[j] * (1.0f / NN) - mu * mu;
    float sc  = gamma[j] * rsqrtf(var + 1e-5f);
    scale[j] = sc;
    shift[j] = beta[j] - mu * sc;
}

// ---------------- h = relu(z*scale + shift + h) ----------------
__global__ __launch_bounds__(256)
void residual_kernel(const float4* __restrict__ z, const float* __restrict__ scale,
                     const float* __restrict__ shift, float4* __restrict__ h, int n4) {
    int i = blockIdx.x * 256 + threadIdx.x;
    if (i >= n4) return;
    int jb = (i & 31) << 2;
    float4 zv = z[i], hv = h[i], o;
    o.x = zv.x * scale[jb + 0] + shift[jb + 0] + hv.x;
    o.y = zv.y * scale[jb + 1] + shift[jb + 1] + hv.y;
    o.z = zv.z * scale[jb + 2] + shift[jb + 2] + hv.z;
    o.w = zv.w * scale[jb + 3] + shift[jb + 3] + hv.w;
    o.x = o.x > 0.f ? o.x : 0.f;  o.y = o.y > 0.f ? o.y : 0.f;
    o.z = o.z > 0.f ? o.z : 0.f;  o.w = o.w > 0.f ? o.w : 0.f;
    h[i] = o;
}

// ---------------- pooling: segment sums into add/cnt ----------------
__global__ __launch_bounds__(128)
void pool_kernel(const float* __restrict__ h, const int* __restrict__ batch,
                 float* __restrict__ add, float* __restrict__ cnt) {
    int n = blockIdx.x, j = threadIdx.x;
    int g = batch[n];
    atomAddF(&add[g * HID + j], h[(long)n * HID + j]);
    if (j == 0) atomAddF(&cnt[g], 1.0f);
}

// ---------------- head MLP (tiny): one block per graph ----------------
__global__ __launch_bounds__(128)
void head_kernel(const float* __restrict__ add, const float* __restrict__ cnt,
                 const float* __restrict__ w0, const float* __restrict__ b0,
                 const float* __restrict__ w1h, const float* __restrict__ b1h,
                 const float* __restrict__ w2h, const float* __restrict__ b2h,
                 float* __restrict__ out) {
    __shared__ float p[256];
    __shared__ float t1[128];
    __shared__ float t2[64];
    int g = blockIdx.x, j = threadIdx.x;
    float c = cnt[g]; c = c > 1.0f ? c : 1.0f;
    float a = add[g * HID + j];
    p[j] = a / c;
    p[128 + j] = a;
    __syncthreads();
    float acc = b0[j];
    for (int k = 0; k < 256; ++k) acc += p[k] * w0[k * 128 + j];
    t1[j] = acc > 0.f ? acc : 0.f;
    __syncthreads();
    if (j < 64) {
        float acc2 = b1h[j];
        for (int k = 0; k < 128; ++k) acc2 += t1[k] * w1h[k * 64 + j];
        t2[j] = acc2 > 0.f ? acc2 : 0.f;
    }
    __syncthreads();
    if (j < 3) {
        float acc3 = b2h[j];
        for (int k = 0; k < 64; ++k) acc3 += t2[k] * w2h[k * 3 + j];
        out[g * 3 + j] = acc3;
    }
}

extern "C" void kernel_launch(void* const* d_in, const int* in_sizes, int n_in,
                              void* d_out, int out_size, void* d_ws, size_t ws_size,
                              hipStream_t stream) {
    const float* x     = (const float*)d_in[0];
    const int*   ei    = (const int*)  d_in[1];
    const float* ea    = (const float*)d_in[2];
    const int*   batch = (const int*)  d_in[3];
    const float* neW = (const float*)d_in[4];
    const float* neB = (const float*)d_in[5];
    const float* eeW = (const float*)d_in[6];
    const float* eeB = (const float*)d_in[7];
    // layers: base 8, stride 7: w1,b1,w2,b2,gamma,beta,eps
    const float* h0W = (const float*)d_in[29];
    const float* h0B = (const float*)d_in[30];
    const float* h1W = (const float*)d_in[31];
    const float* h1B = (const float*)d_in[32];
    const float* h2W = (const float*)d_in[33];
    const float* h2B = (const float*)d_in[34];

    float* ws    = (float*)d_ws;
    float* h     = ws;                       // NN*HID
    float* z     = ws + (size_t)NN * HID;    // NN*HID (in-place MLP)
    float* ssum  = ws + (size_t)2 * NN * HID;
    float* ssum2 = ssum + 128;
    float* scale = ssum2 + 128;
    float* shift = scale + 128;
    float* addp  = shift + 128;              // NG*HID
    float* cntp  = addp + NG * HID;          // NG

    const int n4 = NN * HID / 4;             // 3,200,000
    const int cb = (n4 + 255) / 256;         // 12500

    node_embed_kernel<<<NN, 128, 0, stream>>>(x, neW, neB, h);

    for (int L = 0; L < 3; ++L) {
        const float* w1    = (const float*)d_in[8 + 7 * L + 0];
        const float* b1    = (const float*)d_in[8 + 7 * L + 1];
        const float* w2    = (const float*)d_in[8 + 7 * L + 2];
        const float* b2    = (const float*)d_in[8 + 7 * L + 3];
        const float* gamma = (const float*)d_in[8 + 7 * L + 4];
        const float* beta  = (const float*)d_in[8 + 7 * L + 5];
        const float* epsp  = (const float*)d_in[8 + 7 * L + 6];

        copy_scale_kernel<<<cb, 256, 0, stream>>>((const float4*)h, (float4*)z, epsp, n4);
        zero_kernel<<<1, 256, 0, stream>>>(ssum, 256);   // ssum + ssum2 contiguous
        message_kernel<<<NE / 2, 256, 0, stream>>>(h, ei, ea, eeW, eeB, z);
        gemm_mlp_kernel<<<(NN + 127) / 128, 256, 0, stream>>>(z, w1, b1, w2, b2, ssum, ssum2);
        bn_final_kernel<<<1, 128, 0, stream>>>(ssum, ssum2, gamma, beta, scale, shift);
        residual_kernel<<<cb, 256, 0, stream>>>((const float4*)z, scale, shift, (float4*)h, n4);
    }

    zero_kernel<<<(NG * HID + NG + 255) / 256, 256, 0, stream>>>(addp, NG * HID + NG);
    pool_kernel<<<NN, 128, 0, stream>>>(h, batch, addp, cntp);
    head_kernel<<<NG, 128, 0, stream>>>(addp, cntp, h0W, h0B, h1W, h1B, h2W, h2B,
                                        (float*)d_out);
}
